// PropPredNet_53274774340016
// MI455X (gfx1250) — matrix operand order repaired and checked by reading, not compiled, not run on hardware
//
#include <hip/hip_runtime.h>
#include <hip/hip_bf16.h>

// ---------------------------------------------------------------------------
// EGNN property predictor for MI455X (gfx1250, wave32, WMMA bf16 16x16x32)
// ---------------------------------------------------------------------------
#define B_GRAPH   8
#define N_PROT    2600
#define N_LIG     400
#define N_PER_G   3000            // per-graph nodes
#define N_TOTAL   24000           // B * N_PER_G
#define FP        27
#define FL        13
#define HID       128
#define GS        20              // gaussian bins
#define KNN       32
#define NLAYER    3
#define KPAD      288             // 32 (gauss pad) + 128 + 128
#define EDGES_TOT (N_TOTAL * KNN)

typedef float  v8f   __attribute__((ext_vector_type(8)));
typedef __bf16 v16bf __attribute__((ext_vector_type(16)));

__device__ __forceinline__ unsigned short f2bf(float f) {
    unsigned int u = __float_as_uint(f);
    u += 0x7FFFu + ((u >> 16) & 1u);               // round-to-nearest-even
    return (unsigned short)(u >> 16);
}

// ---------------------------------------------------------------------------
// K0: convert edge-MLP weights into WMMA B-fragment-swizzled bf16 layout.
// B frag element (kc, nt, lane, i):  K = kc*32 + (lane>>4)*8 + (i&7) + (i>>3)*16
//                                    N = nt*16 + (lane&15)
// w1 K-padding: [0,20)=gauss rows, [20,32)=zero, [32,288) = orig rows K-12.
// ---------------------------------------------------------------------------
#define W1F_PER_L (9 * 8 * 32 * 16)   // 36864 elems
#define W2F_PER_L (4 * 8 * 32 * 16)   // 16384 elems
__global__ void wconv_kernel(const float* __restrict__ ew1,
                             const float* __restrict__ ew2,
                             unsigned short* __restrict__ w1f,
                             unsigned short* __restrict__ w2f) {
    int idx = blockIdx.x * 256 + threadIdx.x;
    const int per_l = W1F_PER_L + W2F_PER_L;
    if (idx >= NLAYER * per_l) return;
    int l = idx / per_l;
    int r = idx - l * per_l;
    if (r < W1F_PER_L) {
        int kc = r / 4096, r2 = r % 4096;
        int nt = r2 / 512, r3 = r2 % 512;
        int lane = r3 / 16, i = r3 % 16;
        int K = kc * 32 + ((lane >> 4) << 3) + (i & 7) + ((i >> 3) << 4);
        int N = nt * 16 + (lane & 15);
        float val = 0.f;
        if (K < GS)       val = ew1[(l * 276 + K) * HID + N];
        else if (K >= 32) val = ew1[(l * 276 + (K - 12)) * HID + N];
        w1f[l * W1F_PER_L + r] = f2bf(val);
    } else {
        int r1 = r - W1F_PER_L;
        int kc = r1 / 4096, r2 = r1 % 4096;
        int nt = r2 / 512, r3 = r2 % 512;
        int lane = r3 / 16, i = r3 % 16;
        int K = kc * 32 + ((lane >> 4) << 3) + (i & 7) + ((i >> 3) << 4);
        int N = nt * 16 + (lane & 15);
        w2f[l * W2F_PER_L + r1] = f2bf(ew2[(l * HID + K) * HID + N]);
    }
}

// ---------------------------------------------------------------------------
// K1: atom embeddings + compose_context (closed-form stable-sort permutation).
// Node i: graph b=i/3000, rank r=i%3000; r<2600 -> protein row, else ligand.
// ---------------------------------------------------------------------------
__global__ void embed_kernel(const float* __restrict__ ppos,
                             const float* __restrict__ pfeat,
                             const float* __restrict__ lpos,
                             const float* __restrict__ lfeat,
                             const float* __restrict__ pw, const float* __restrict__ pb,
                             const float* __restrict__ lw, const float* __restrict__ lb,
                             float* __restrict__ h, unsigned short* __restrict__ hbf,
                             float* __restrict__ pos) {
    __shared__ float fs[32];
    int i = blockIdx.x;
    int b = i / N_PER_G;
    int r = i - b * N_PER_G;
    int F; const float* feat; const float* W; const float* bias; const float* sp;
    if (r < N_PROT) {
        int srow = b * N_PROT + r;
        F = FP; feat = pfeat + (long)srow * FP; W = pw; bias = pb; sp = ppos + (long)srow * 3;
    } else {
        int srow = b * N_LIG + (r - N_PROT);
        F = FL; feat = lfeat + (long)srow * FL; W = lw; bias = lb; sp = lpos + (long)srow * 3;
    }
    int c = threadIdx.x;
    if (c < F) fs[c] = feat[c];
    __syncthreads();
    float acc = bias[c];
    for (int k = 0; k < F; ++k) acc += fs[k] * W[k * HID + c];
    long gi = (long)i * HID + c;
    h[gi] = acc;
    hbf[gi] = f2bf(acc);
    if (c < 3) pos[(long)i * 3 + c] = sp[c];
}

// ---------------------------------------------------------------------------
// K2: per-graph kNN (K=32, no self loops, ties by smaller index like top_k).
// One block = 64 nodes of one graph; graph pos + 8 per-wave dist caches in LDS.
// Each lane owns a 32-strided slice of the dist array -> no cross-lane LDS.
// ---------------------------------------------------------------------------
#define CHUNKS_PER_G 47           // ceil(3000/64)
__global__ void knn_kernel(const float* __restrict__ pos, int* __restrict__ nbr) {
    extern __shared__ __align__(32) unsigned char dynsm_knn[];
    float* px   = (float*)dynsm_knn;
    float* py   = px + N_PER_G;
    float* pz   = py + N_PER_G;
    float* dist = pz + N_PER_G;                        // [8][3000]
    int b = blockIdx.x / CHUNKS_PER_G;
    int chunk = blockIdx.x - b * CHUNKS_PER_G;
    int gbase = b * N_PER_G;
    for (int k = threadIdx.x; k < N_PER_G; k += 256) {
        px[k] = pos[(long)(gbase + k) * 3 + 0];
        py[k] = pos[(long)(gbase + k) * 3 + 1];
        pz[k] = pos[(long)(gbase + k) * 3 + 2];
    }
    __syncthreads();
    int w = threadIdx.x >> 5, lane = threadIdx.x & 31;
    float* dw = dist + w * N_PER_G;
    for (int s = 0; s < 8; ++s) {
        int j = chunk * 64 + w * 8 + s;                // uniform per wave
        if (j >= N_PER_G) continue;
        float xj = px[j], yj = py[j], zj = pz[j];
        for (int c = lane; c < N_PER_G; c += 32) {
            float dx = px[c] - xj, dy = py[c] - yj, dz = pz[c] - zj;
            dw[c] = (c == j) ? 1e30f : (dx * dx + dy * dy + dz * dz);
        }
        for (int k = 0; k < KNN; ++k) {
            float bd = 1e30f; int bi = 0x7fffffff;
            for (int c = lane; c < N_PER_G; c += 32) {
                float d = dw[c];
                if (d < bd || (d == bd && c < bi)) { bd = d; bi = c; }
            }
            #pragma unroll
            for (int m = 16; m >= 1; m >>= 1) {
                float od = __shfl_xor(bd, m, 32);
                int   oi = __shfl_xor(bi, m, 32);
                if (od < bd || (od == bd && oi < bi)) { bd = od; bi = oi; }
            }
            if (lane == (bi & 31)) dw[bi] = 1e30f;     // exclude winner
            if (lane == 0) nbr[(long)(gbase + j) * KNN + k] = gbase + bi;
        }
    }
}

// ---------------------------------------------------------------------------
// K3: fused EGNN layer. Block = 4 dst nodes = 128 edges; 8 waves; wave w owns
// M-tile w. GEMM1 [128 x 288] x [288 x 128] -> relu -> GEMM2 [.x128]x[128x128]
// -> relu -> gate -> LDS scatter -> node MLP + residual.
// A fragment layout (16-bit A 16x32, ISA 7.12.2):
//   m -> lane = (m&15) + 16*((K>>3)&1);  i = (K&7) + 8*((K>>4)&1); kc = K>>5
// ---------------------------------------------------------------------------
__global__ __launch_bounds__(256)
void egnn_layer_kernel(const float* __restrict__ pos,
                       const float* __restrict__ h_cur,
                       const unsigned short* __restrict__ hbf_cur,
                       const int* __restrict__ nbr,
                       const unsigned short* __restrict__ w1f,
                       const unsigned short* __restrict__ w2f,
                       const float* __restrict__ eb1, const float* __restrict__ eb2,
                       const float* __restrict__ infw, const float* __restrict__ infb,
                       const float* __restrict__ nw1, const float* __restrict__ nb1,
                       const float* __restrict__ nw2, const float* __restrict__ nb2,
                       float* __restrict__ h_next, unsigned short* __restrict__ hbf_next) {
    __shared__ float mi_s[4 * HID];
    __shared__ float hrow_s[4 * HID];
    __shared__ float hid_s[4 * HID];
    __shared__ float elen_s[128];
    __shared__ int   src_s[128];
    __shared__ float eb1_s[HID], eb2_s[HID], infw_s[HID];
    extern __shared__ __align__(32) unsigned char dynsm[];
    unsigned short* a_frag = (unsigned short*)dynsm;            // [8][9][32][16]
    unsigned short* m_frag = (unsigned short*)(dynsm + 8 * 9 * 512 * 2); // [8][4][32][16]

    const int tid = threadIdx.x, lane = tid & 31, w = tid >> 5;
    const int dstbase = blockIdx.x * 4;
    const long ebase = (long)blockIdx.x * 128;

    // ---- preamble: edge srcs, lengths, small params, dst h rows ----
    if (tid < 128) {
        int s = nbr[ebase + tid];
        src_s[tid] = s;
        int dn = dstbase + (tid >> 5);
        float dx = pos[(long)dn * 3 + 0] - pos[(long)s * 3 + 0];
        float dy = pos[(long)dn * 3 + 1] - pos[(long)s * 3 + 1];
        float dz = pos[(long)dn * 3 + 2] - pos[(long)s * 3 + 2];
        elen_s[tid] = sqrtf(dx * dx + dy * dy + dz * dz);
        eb1_s[tid] = eb1[tid];
        eb2_s[tid] = eb2[tid];
        infw_s[tid] = infw[tid];
    }
    for (int k = tid; k < 4 * HID; k += 256) {
        mi_s[k] = 0.f;
        hrow_s[k] = h_cur[(long)(dstbase + (k >> 7)) * HID + (k & 127)];
    }
    __syncthreads();

    // ---- stage X (128 x 288 bf16) directly in A-fragment layout ----
    const float step = 10.0f / (GS - 1);
    const float coeff = -0.5f / (step * step);
    for (int t = tid; t < 128 * 36; t += 256) {
        int m = t / 36, g = t - m * 36;                 // g: 8-wide K group
        int mt = m >> 4;
        int lane2 = (m & 15) + ((g & 1) << 4);
        int i0 = ((g >> 1) & 1) << 3;
        int kc = g >> 2;
        unsigned short* dst = &a_frag[((((mt * 9 + kc) << 5) + lane2) << 4) + i0];
        if (g < 4) {                                    // gaussian smearing cols
            float el = elen_s[m];
            union { unsigned short v[8]; uint4 q; } u;
            int K0 = g * 8;
            #pragma unroll
            for (int j = 0; j < 8; ++j) {
                int col = K0 + j;
                float val = 0.f;
                if (col < GS) { float d = el - col * step; val = __expf(coeff * d * d); }
                u.v[j] = f2bf(val);
            }
            *(uint4*)dst = u.q;
        } else {
            int c0  = (g < 20) ? (g * 8 - 32) : (g * 8 - 160);
            int row = (g < 20) ? (dstbase + (m >> 5)) : src_s[m];
            *(uint4*)dst = *(const uint4*)(hbf_cur + (long)row * HID + c0);
        }
    }
    __syncthreads();

    // ---- GEMM1: 16x288 @ 288x128 per wave (72 WMMAs), C init = bias ----
    const int n15 = lane & 15;
    v8f c[8];
    #pragma unroll
    for (int t = 0; t < 8; ++t) {
        float bv = eb1_s[t * 16 + n15];
        #pragma unroll
        for (int q = 0; q < 8; ++q) c[t][q] = bv;
    }
    for (int kc = 0; kc < 9; ++kc) {
        v16bf a = *(const v16bf*)&a_frag[(((w * 9 + kc) << 5) + lane) << 4];
        #pragma unroll
        for (int t = 0; t < 8; ++t) {
            v16bf bm = *(const v16bf*)&w1f[(((kc * 8 + t) << 5) + lane) << 4];
            c[t] = __builtin_amdgcn_wmma_f32_16x16x32_bf16(false, a, false, bm,
                                                           (short)0, c[t], false, false);
        }
    }
    // relu -> bf16 -> wave-local A fragments for GEMM2
    {
        int kid_hi = ((lane >> 4) << 3);
        int lhib = (n15 >> 3) << 4;
        #pragma unroll
        for (int t = 0; t < 8; ++t) {
            int kcd = t >> 1;
            int i = (n15 & 7) + ((t & 1) << 3);
            #pragma unroll
            for (int v2 = 0; v2 < 8; ++v2) {
                float x = c[t][v2]; x = x > 0.f ? x : 0.f;
                int lp = (v2 + kid_hi) + lhib;
                m_frag[((((w << 2) + kcd) << 5) + lp) * 16 + i] = f2bf(x);
            }
        }
    }
    // ---- GEMM2: 16x128 @ 128x128 per wave (32 WMMAs) ----
    v8f c2[8];
    #pragma unroll
    for (int t = 0; t < 8; ++t) {
        float bv = eb2_s[t * 16 + n15];
        #pragma unroll
        for (int q = 0; q < 8; ++q) c2[t][q] = bv;
    }
    for (int kc = 0; kc < 4; ++kc) {
        v16bf a = *(const v16bf*)&m_frag[((((w << 2) + kc) << 5) + lane) << 4];
        #pragma unroll
        for (int t = 0; t < 8; ++t) {
            v16bf bm = *(const v16bf*)&w2f[(((kc * 8 + t) << 5) + lane) << 4];
            c2[t] = __builtin_amdgcn_wmma_f32_16x16x32_bf16(false, a, false, bm,
                                                            (short)0, c2[t], false, false);
        }
    }
    #pragma unroll
    for (int t = 0; t < 8; ++t)
        #pragma unroll
        for (int q = 0; q < 8; ++q) c2[t][q] = c2[t][q] > 0.f ? c2[t][q] : 0.f;

    // ---- gate: e = sigmoid(m . inf_w + b) per edge row, then scatter m*e ----
    float partial[8] = {0, 0, 0, 0, 0, 0, 0, 0};
    #pragma unroll
    for (int t = 0; t < 8; ++t) {
        float wv = infw_s[t * 16 + n15];
        #pragma unroll
        for (int v2 = 0; v2 < 8; ++v2) partial[v2] += c2[t][v2] * wv;
    }
    float ib = infb[0];
    float e8[8];
    #pragma unroll
    for (int v2 = 0; v2 < 8; ++v2) {
        float s = partial[v2];
        s += __shfl_xor(s, 1, 32); s += __shfl_xor(s, 2, 32);
        s += __shfl_xor(s, 4, 32); s += __shfl_xor(s, 8, 32);   // reduce within half
        e8[v2] = 1.f / (1.f + __expf(-(s + ib)));
    }
    int dl = w >> 1;                                            // local dst node
    #pragma unroll
    for (int t = 0; t < 8; ++t) {
        float cs = 0.f;
        #pragma unroll
        for (int v2 = 0; v2 < 8; ++v2) cs += c2[t][v2] * e8[v2];
        cs += __shfl_xor(cs, 16, 32);                            // both halves
        if (lane < 16) atomicAdd(&mi_s[dl * HID + t * 16 + lane], cs);
    }
    __syncthreads();

    // ---- node MLP: z=[mi,h]; o = relu(z@nw1+nb1)@nw2+nb2; h' = h + o ----
    for (int o = tid; o < 4 * HID; o += 256) {
        int node = o >> 7, j = o & 127;
        float acc = nb1[j];
        const float* z1 = &mi_s[node * HID];
        const float* z2 = &hrow_s[node * HID];
        for (int k = 0; k < HID; ++k) acc += z1[k] * nw1[k * HID + j];
        for (int k = 0; k < HID; ++k) acc += z2[k] * nw1[(HID + k) * HID + j];
        hid_s[o] = acc > 0.f ? acc : 0.f;
    }
    __syncthreads();
    for (int o = tid; o < 4 * HID; o += 256) {
        int node = o >> 7, j = o & 127;
        float acc = nb2[j];
        const float* hd = &hid_s[node * HID];
        for (int k = 0; k < HID; ++k) acc += hd[k] * nw2[k * HID + j];
        float hn = hrow_s[o] + acc;
        long gi = (long)(dstbase + node) * HID + j;
        h_next[gi] = hn;
        hbf_next[gi] = f2bf(hn);
    }
}

// ---------------------------------------------------------------------------
// K4: per-graph pooling (segment_sum over 3000 nodes)
// ---------------------------------------------------------------------------
__global__ void pool_kernel(const float* __restrict__ h, float* __restrict__ pooled) {
    int b = blockIdx.x, c = threadIdx.x;
    float s = 0.f;
    const float* hb = h + (long)b * N_PER_G * HID + c;
    for (int j = 0; j < N_PER_G; ++j) s += hb[(long)j * HID];
    pooled[b * HID + c] = s;
}

// ---------------------------------------------------------------------------
// K5: out_block + output_kind mask-select
// ---------------------------------------------------------------------------
__global__ void out_kernel(const float* __restrict__ pooled,
                           const float* __restrict__ ow1, const float* __restrict__ ob1,
                           const float* __restrict__ ow2, const float* __restrict__ ob2,
                           const int* __restrict__ okind, float* __restrict__ dout) {
    __shared__ float ps[B_GRAPH * HID];
    __shared__ float us[B_GRAPH * HID];
    int tid = threadIdx.x;
    for (int k = tid; k < B_GRAPH * HID; k += 256) ps[k] = pooled[k];
    __syncthreads();
    for (int o = tid; o < B_GRAPH * HID; o += 256) {
        int b = o >> 7, j = o & 127;
        float acc = ob1[j];
        for (int k = 0; k < HID; ++k) acc += ps[b * HID + k] * ow1[k * HID + j];
        float sp = (acc > 20.f) ? acc : log1pf(__expf(acc));
        us[o] = sp - 0.69314718055994531f;               // ShiftedSoftplus
    }
    __syncthreads();
    if (tid < B_GRAPH) {
        int b = tid;
        int kind = okind[b] - 1;
        float acc = ob2[kind];
        for (int j = 0; j < HID; ++j) acc += us[b * HID + j] * ow2[j * 3 + kind];
        dout[b] = acc;
    }
}

// ---------------------------------------------------------------------------
// host: workspace carve + launches
// ---------------------------------------------------------------------------
extern "C" void kernel_launch(void* const* d_in, const int* in_sizes, int n_in,
                              void* d_out, int out_size, void* d_ws, size_t ws_size,
                              hipStream_t stream) {
    const float* ppos  = (const float*)d_in[0];
    const float* pfeat = (const float*)d_in[1];
    const float* lpos  = (const float*)d_in[2];
    const float* lfeat = (const float*)d_in[3];
    // d_in[4], d_in[5]: batch ids (permutation is closed-form; not needed)
    const int*   okind = (const int*)d_in[6];
    const float* pw  = (const float*)d_in[7];
    const float* pb  = (const float*)d_in[8];
    const float* lw  = (const float*)d_in[9];
    const float* lb  = (const float*)d_in[10];
    const float* ew1 = (const float*)d_in[11];
    const float* eb1 = (const float*)d_in[12];
    const float* ew2 = (const float*)d_in[13];
    const float* eb2 = (const float*)d_in[14];
    const float* ifw = (const float*)d_in[15];
    const float* ifb = (const float*)d_in[16];
    const float* nw1 = (const float*)d_in[17];
    const float* nb1 = (const float*)d_in[18];
    const float* nw2 = (const float*)d_in[19];
    const float* nb2 = (const float*)d_in[20];
    const float* ow1 = (const float*)d_in[21];
    const float* ob1 = (const float*)d_in[22];
    const float* ow2 = (const float*)d_in[23];
    const float* ob2 = (const float*)d_in[24];

    char* ws = (char*)d_ws;
    size_t off = 0;
    float* pos = (float*)(ws + off);          off += (size_t)N_TOTAL * 3 * 4;      // 288000
    float* hA  = (float*)(ws + off);          off += (size_t)N_TOTAL * HID * 4;    // 12.3MB
    float* hB  = (float*)(ws + off);          off += (size_t)N_TOTAL * HID * 4;
    unsigned short* hbA = (unsigned short*)(ws + off); off += (size_t)N_TOTAL * HID * 2;
    unsigned short* hbB = (unsigned short*)(ws + off); off += (size_t)N_TOTAL * HID * 2;
    int*   nbr = (int*)(ws + off);            off += (size_t)EDGES_TOT * 4;        // 3.07MB
    unsigned short* w1f = (unsigned short*)(ws + off); off += (size_t)NLAYER * W1F_PER_L * 2;
    unsigned short* w2f = (unsigned short*)(ws + off); off += (size_t)NLAYER * W2F_PER_L * 2;
    off = (off + 255) & ~(size_t)255;
    float* pooled = (float*)(ws + off);       off += (size_t)B_GRAPH * HID * 4;
    (void)ws_size; (void)in_sizes; (void)n_in; (void)out_size;

    // K0: swizzle edge weights to WMMA fragment layout (bf16)
    {
        int total = NLAYER * (W1F_PER_L + W2F_PER_L);
        wconv_kernel<<<(total + 255) / 256, 256, 0, stream>>>(ew1, ew2, w1f, w2f);
    }
    // K1: embeddings + compose (sorted order), f32 + bf16 h, sorted pos
    embed_kernel<<<N_TOTAL, HID, 0, stream>>>(ppos, pfeat, lpos, lfeat,
                                              pw, pb, lw, lb, hA, hbA, pos);
    // K2: kNN graph (LDS-cached positions + per-wave distance arrays)
    {
        size_t shm = (size_t)(3 * N_PER_G + 8 * N_PER_G) * 4;   // 132 KB
        knn_kernel<<<B_GRAPH * CHUNKS_PER_G, 256, shm, stream>>>(pos, nbr);
    }
    // K3: three fused EGNN layers (ping-pong h)
    {
        size_t shm = (size_t)(8 * 9 * 512 + 8 * 4 * 512) * 2;   // 104 KB dynamic
        float* hc = hA; unsigned short* hbc = hbA;
        float* hn = hB; unsigned short* hbn = hbB;
        for (int l = 0; l < NLAYER; ++l) {
            egnn_layer_kernel<<<N_TOTAL / 4, 256, shm, stream>>>(
                pos, hc, hbc, nbr,
                w1f + (size_t)l * W1F_PER_L, w2f + (size_t)l * W2F_PER_L,
                eb1 + l * HID, eb2 + l * HID, ifw + l * HID, ifb + l,
                nw1 + (size_t)l * 2 * HID * HID, nb1 + l * HID,
                nw2 + (size_t)l * HID * HID, nb2 + l * HID,
                hn, hbn);
            float* tf = hc; hc = hn; hn = tf;
            unsigned short* tb = hbc; hbc = hbn; hbn = tb;
        }
        // after 3 layers, hc holds the final h
        pool_kernel<<<B_GRAPH, HID, 0, stream>>>(hc, pooled);
    }
    // K5: out block + kind select
    out_kernel<<<1, 256, 0, stream>>>(pooled, ow1, ob1, ow2, ob2, okind, (float*)d_out);
}
